// SimpleMoELayer_21947282883176
// MI455X (gfx1250) — compile-verified, matrix operand order checked by reading, not run on hardware
//
#include <hip/hip_runtime.h>

// ---------------------------------------------------------------------------
// MoE (top-1 routed) for gfx1250: bf16 WMMA grouped GEMM
//   + async global->LDS copies for bf16 A tiles (ASYNCcnt pipeline)
//   + double-buffered A, register-staged B (fp32->bf16 inline conversion)
// B=2, S=2048 -> T=4096 tokens, H=2048, I=1408, E=8
// ---------------------------------------------------------------------------

#define T_TOK 4096
#define H_DIM 2048
#define I_DIM 1408
#define E_NUM 8
#define LDS_STRIDE 72   // 64 + 8 bf16 pad to spread LDS banks

typedef __attribute__((ext_vector_type(16))) __bf16        v16bf;
typedef __attribute__((ext_vector_type(8)))  float         v8f;
typedef __attribute__((ext_vector_type(4)))  float         f32x4;
typedef __attribute__((ext_vector_type(4)))  unsigned int  u32x4;
typedef __attribute__((ext_vector_type(2)))  unsigned int  u32x2;

union FragBF { u32x4 q[2]; v16bf v; };

// Async copy of 16 bytes global(bf16) -> LDS, tracked by ASYNCcnt.
__device__ __forceinline__ void async_copy_b128(void* lds_ptr, const void* gptr) {
  unsigned lds = (unsigned)(unsigned long long)lds_ptr;  // low 32 bits = LDS offset
  asm volatile("global_load_async_to_lds_b128 %0, %1, off"
               :: "v"(lds), "v"(gptr) : "memory");
}
__device__ __forceinline__ void wait_async0() {
  asm volatile("s_wait_asynccnt 0x0" ::: "memory");
}

// ---------------- workspace layout (bytes) ----------------
#define WS_COUNTS     0
#define WS_CURSOR     64
#define WS_OFFS       128
#define WS_EXPERT     256
#define WS_WTOK       (WS_EXPERT + T_TOK * 4)
#define WS_PERM       (WS_WTOK   + T_TOK * 4)
#define WS_WPACK      (WS_PERM   + T_TOK * 4)
#define WS_XPACK      (WS_WPACK  + T_TOK * 4)                    // bf16 T*H
#define WS_ACTPACK    (WS_XPACK  + (size_t)T_TOK * H_DIM * 2)    // bf16 T*I

// ---------------------------------------------------------------------------
__global__ void moe_init(int* __restrict__ counts, int* __restrict__ cursor) {
  int i = threadIdx.x;
  if (i < E_NUM) { counts[i] = 0; cursor[i] = 0; }
}

// ---------------------------------------------------------------------------
__global__ void moe_router(const float* __restrict__ x,
                           const float* __restrict__ rw,
                           int* __restrict__ expert_id,
                           float* __restrict__ wtok,
                           int* __restrict__ counts) {
  int t = blockIdx.x * blockDim.x + threadIdx.x;
  if (t >= T_TOK) return;
  float acc[E_NUM];
#pragma unroll
  for (int e = 0; e < E_NUM; ++e) acc[e] = 0.f;
  const float* xr = x + (size_t)t * H_DIM;
  for (int h = 0; h < H_DIM; h += 4) {
    f32x4 xv = *(const f32x4*)(xr + h);
#pragma unroll
    for (int e = 0; e < E_NUM; ++e) {
      f32x4 rv = *(const f32x4*)(rw + (size_t)e * H_DIM + h);
      acc[e] = __builtin_fmaf(xv.x, rv.x, acc[e]);
      acc[e] = __builtin_fmaf(xv.y, rv.y, acc[e]);
      acc[e] = __builtin_fmaf(xv.z, rv.z, acc[e]);
      acc[e] = __builtin_fmaf(xv.w, rv.w, acc[e]);
    }
  }
  float v0 = acc[0]; int i0 = 0; float v1 = -3.4e38f;
#pragma unroll
  for (int e = 1; e < E_NUM; ++e) {
    if (acc[e] > v0) { v1 = v0; v0 = acc[e]; i0 = e; }
    else if (acc[e] > v1) { v1 = acc[e]; }
  }
  float w = 1.f / (1.f + __expf(v1 - v0));
  expert_id[t] = i0;
  wtok[t] = w;
  atomicAdd(&counts[i0], 1);
}

// ---------------------------------------------------------------------------
__global__ void moe_prefix(const int* __restrict__ counts, int* __restrict__ offs) {
  if (threadIdx.x == 0) {
    int s = 0;
#pragma unroll
    for (int e = 0; e < E_NUM; ++e) { offs[e] = s; s += counts[e]; }
    offs[E_NUM] = s;
  }
}

// ---------------------------------------------------------------------------
__global__ void moe_scatter(const int* __restrict__ expert_id,
                            const float* __restrict__ wtok,
                            const int* __restrict__ offs,
                            int* __restrict__ cursor,
                            int* __restrict__ perm,
                            float* __restrict__ wpacked) {
  int t = blockIdx.x * blockDim.x + threadIdx.x;
  if (t >= T_TOK) return;
  int e = expert_id[t];
  int pos = offs[e] + atomicAdd(&cursor[e], 1);
  perm[pos] = t;
  wpacked[pos] = wtok[t];
}

// ---------------------------------------------------------------------------
__global__ void moe_gather(const float* __restrict__ x,
                           const int* __restrict__ perm,
                           __bf16* __restrict__ xpack) {
  int p = blockIdx.x;
  int t = perm[p];
  const f32x4* src = (const f32x4*)(x + (size_t)t * H_DIM);
  __bf16* dst = xpack + (size_t)p * H_DIM;
  int i = threadIdx.x;
  f32x4 a = src[i * 2];
  f32x4 b = src[i * 2 + 1];
  __bf16 tmp[8] __attribute__((aligned(16))) = {
      (__bf16)a.x, (__bf16)a.y, (__bf16)a.z, (__bf16)a.w,
      (__bf16)b.x, (__bf16)b.y, (__bf16)b.z, (__bf16)b.w};
  *(u32x4*)(dst + i * 8) = *(const u32x4*)tmp;
}

// ---------------------------------------------------------------------------
// GEMM1: act = silu(x@gate^T) * (x@up^T). Tile 128(M) x 64(N over I), K=H.
// A double-buffered via async-to-LDS; gate/up weights reg-staged fp32->bf16.
__global__ __launch_bounds__(256) void moe_gemm1(
    const __bf16* __restrict__ xpack,
    const float* __restrict__ gate_w,
    const float* __restrict__ up_w,
    const int* __restrict__ offs,
    __bf16* __restrict__ actpack) {
  __shared__ __bf16 sA [2][128 * LDS_STRIDE];
  __shared__ __bf16 sBg[64 * LDS_STRIDE];
  __shared__ __bf16 sBu[64 * LDS_STRIDE];

  const int e    = blockIdx.z;
  const int base = offs[e];
  const int cnt  = offs[e + 1] - base;
  const int m0   = blockIdx.y * 128;
  if (m0 >= cnt) return;
  const int rem = cnt - m0;
  const int n0  = blockIdx.x * 64;

  const int tid  = threadIdx.x;
  const int lane = tid & 31;
  const int wave = tid >> 5;
  const int wm   = (wave & 3) * 32;
  const int wn   = (wave >> 2) * 32;

  v8f accg[2][2] = {};
  v8f accu[2][2] = {};

  const size_t wbase = (size_t)e * I_DIM * H_DIM;
  const float* gw = gate_w + wbase + (size_t)n0 * H_DIM;
  const float* uw = up_w   + wbase + (size_t)n0 * H_DIM;

  f32x4 gReg[4], uReg[4];

  auto gloadB = [&](int kc) {
#pragma unroll
    for (int i = 0; i < 4; ++i) {
      int idx = tid + i * 256;
      int row = idx >> 4;
      int col = (idx & 15) * 4;
      gReg[i] = *(const f32x4*)(gw + (size_t)row * H_DIM + kc + col);
      uReg[i] = *(const f32x4*)(uw + (size_t)row * H_DIM + kc + col);
    }
  };
  auto storeB = [&]() {
#pragma unroll
    for (int i = 0; i < 4; ++i) {
      int idx = tid + i * 256;
      int row = idx >> 4;
      int col = (idx & 15) * 4;
      __bf16 gb[4] __attribute__((aligned(8))) =
          {(__bf16)gReg[i].x, (__bf16)gReg[i].y, (__bf16)gReg[i].z, (__bf16)gReg[i].w};
      __bf16 ub[4] __attribute__((aligned(8))) =
          {(__bf16)uReg[i].x, (__bf16)uReg[i].y, (__bf16)uReg[i].z, (__bf16)uReg[i].w};
      *(u32x2*)(&sBg[row * LDS_STRIDE + col]) = *(const u32x2*)gb;
      *(u32x2*)(&sBu[row * LDS_STRIDE + col]) = *(const u32x2*)ub;
    }
  };
  auto stageA = [&](int buf, int kc) {  // async copy, bf16 rows, pad pre-zeroed
#pragma unroll
    for (int i = 0; i < 4; ++i) {
      int idx = tid + i * 256;
      int row = idx >> 3;
      int col = (idx & 7) * 8;
      if (row < rem)
        async_copy_b128(&sA[buf][row * LDS_STRIDE + col],
                        xpack + (size_t)(base + m0 + row) * H_DIM + kc + col);
    }
  };

  // prologue: zero-fill pad rows once (async never touches them)
#pragma unroll
  for (int i = 0; i < 4; ++i) {
    int idx = tid + i * 256;
    int row = idx >> 3;
    int col = (idx & 7) * 8;
    if (row >= rem) {
      u32x4 z = {};
      *(u32x4*)(&sA[0][row * LDS_STRIDE + col]) = z;
      *(u32x4*)(&sA[1][row * LDS_STRIDE + col]) = z;
    }
  }
  gloadB(0);
  stageA(0, 0);

  const int NK = H_DIM / 64;
  for (int k = 0; k < NK; ++k) {
    const int cur = k & 1;
    wait_async0();            // A(cur) resident in LDS
    __syncthreads();          // prior compute reads drained; A visible to all
    storeB();                 // B(k): regs -> bf16 -> LDS
    if (k + 1 < NK) {
      gloadB((k + 1) * 64);               // overlap global latency with WMMAs
      stageA(cur ^ 1, (k + 1) * 64);      // async A next chunk, other buffer
    }
    __syncthreads();          // B(k) visible

    const int mloc = lane & 15;
    const int kh   = (lane >> 4) * 8;
#pragma unroll
    for (int kk = 0; kk < 64; kk += 32) {
      FragBF a[2], bg[2], bu[2];
#pragma unroll
      for (int s = 0; s < 2; ++s) {
        const __bf16* ap = &sA[cur][(wm + s * 16 + mloc) * LDS_STRIDE + kk + kh];
        a[s].q[0]  = *(const u32x4*)(ap);
        a[s].q[1]  = *(const u32x4*)(ap + 16);
        const __bf16* gp = &sBg[(wn + s * 16 + mloc) * LDS_STRIDE + kk + kh];
        bg[s].q[0] = *(const u32x4*)(gp);
        bg[s].q[1] = *(const u32x4*)(gp + 16);
        const __bf16* up = &sBu[(wn + s * 16 + mloc) * LDS_STRIDE + kk + kh];
        bu[s].q[0] = *(const u32x4*)(up);
        bu[s].q[1] = *(const u32x4*)(up + 16);
      }
#pragma unroll
      for (int ms = 0; ms < 2; ++ms)
#pragma unroll
        for (int ns = 0; ns < 2; ++ns) {
          accg[ms][ns] = __builtin_amdgcn_wmma_f32_16x16x32_bf16(
              false, a[ms].v, false, bg[ns].v, (short)0, accg[ms][ns], false, false);
          accu[ms][ns] = __builtin_amdgcn_wmma_f32_16x16x32_bf16(
              false, a[ms].v, false, bu[ns].v, (short)0, accu[ms][ns], false, false);
        }
    }
  }

  const int ncol = lane & 15;
  const int mhi  = (lane >> 4) * 8;
#pragma unroll
  for (int ms = 0; ms < 2; ++ms)
#pragma unroll
    for (int ns = 0; ns < 2; ++ns)
#pragma unroll
      for (int vr = 0; vr < 8; ++vr) {
        int mrow = wm + ms * 16 + mhi + vr;
        if (mrow < rem) {
          float g = accg[ms][ns][vr];
          float u = accu[ms][ns][vr];
          float act = (g / (1.f + __expf(-g))) * u;
          int icol = n0 + wn + ns * 16 + ncol;
          actpack[(size_t)(base + m0 + mrow) * I_DIM + icol] = (__bf16)act;
        }
      }
}

// ---------------------------------------------------------------------------
// GEMM2: out[t] = weight[t] * (act @ down^T). Tile 128(M) x 128(N over H), K=I.
__global__ __launch_bounds__(256) void moe_gemm2(
    const __bf16* __restrict__ actpack,
    const float* __restrict__ down_w,
    const int* __restrict__ offs,
    const int* __restrict__ perm,
    const float* __restrict__ wpacked,
    float* __restrict__ out) {
  __shared__ __bf16 sA[2][128 * LDS_STRIDE];
  __shared__ __bf16 sB[128 * LDS_STRIDE];

  const int e    = blockIdx.z;
  const int base = offs[e];
  const int cnt  = offs[e + 1] - base;
  const int m0   = blockIdx.y * 128;
  if (m0 >= cnt) return;
  const int rem = cnt - m0;
  const int n0  = blockIdx.x * 128;

  const int tid  = threadIdx.x;
  const int lane = tid & 31;
  const int wave = tid >> 5;
  const int wm   = (wave & 3) * 32;
  const int wn   = (wave >> 2) * 64;

  v8f acc[2][4] = {};

  const float* dw = down_w + (size_t)e * H_DIM * I_DIM + (size_t)n0 * I_DIM;

  f32x4 dReg[8];
  auto gloadB = [&](int kc) {
#pragma unroll
    for (int i = 0; i < 8; ++i) {
      int idx = tid + i * 256;
      int row = idx >> 4;
      int col = (idx & 15) * 4;
      dReg[i] = *(const f32x4*)(dw + (size_t)row * I_DIM + kc + col);
    }
  };
  auto storeB = [&]() {
#pragma unroll
    for (int i = 0; i < 8; ++i) {
      int idx = tid + i * 256;
      int row = idx >> 4;
      int col = (idx & 15) * 4;
      __bf16 db[4] __attribute__((aligned(8))) =
          {(__bf16)dReg[i].x, (__bf16)dReg[i].y, (__bf16)dReg[i].z, (__bf16)dReg[i].w};
      *(u32x2*)(&sB[row * LDS_STRIDE + col]) = *(const u32x2*)db;
    }
  };
  auto stageA = [&](int buf, int kc) {
#pragma unroll
    for (int i = 0; i < 4; ++i) {
      int idx = tid + i * 256;
      int row = idx >> 3;
      int col = (idx & 7) * 8;
      if (row < rem)
        async_copy_b128(&sA[buf][row * LDS_STRIDE + col],
                        actpack + (size_t)(base + m0 + row) * I_DIM + kc + col);
    }
  };

#pragma unroll
  for (int i = 0; i < 4; ++i) {
    int idx = tid + i * 256;
    int row = idx >> 3;
    int col = (idx & 7) * 8;
    if (row >= rem) {
      u32x4 z = {};
      *(u32x4*)(&sA[0][row * LDS_STRIDE + col]) = z;
      *(u32x4*)(&sA[1][row * LDS_STRIDE + col]) = z;
    }
  }
  gloadB(0);
  stageA(0, 0);

  const int NK = I_DIM / 64;
  for (int k = 0; k < NK; ++k) {
    const int cur = k & 1;
    wait_async0();
    __syncthreads();
    storeB();
    if (k + 1 < NK) {
      gloadB((k + 1) * 64);
      stageA(cur ^ 1, (k + 1) * 64);
    }
    __syncthreads();

    const int mloc = lane & 15;
    const int kh   = (lane >> 4) * 8;
#pragma unroll
    for (int kk = 0; kk < 64; kk += 32) {
      FragBF a[2], b[4];
#pragma unroll
      for (int s = 0; s < 2; ++s) {
        const __bf16* ap = &sA[cur][(wm + s * 16 + mloc) * LDS_STRIDE + kk + kh];
        a[s].q[0] = *(const u32x4*)(ap);
        a[s].q[1] = *(const u32x4*)(ap + 16);
      }
#pragma unroll
      for (int s = 0; s < 4; ++s) {
        const __bf16* bp = &sB[(wn + s * 16 + mloc) * LDS_STRIDE + kk + kh];
        b[s].q[0] = *(const u32x4*)(bp);
        b[s].q[1] = *(const u32x4*)(bp + 16);
      }
#pragma unroll
      for (int ms = 0; ms < 2; ++ms)
#pragma unroll
        for (int ns = 0; ns < 4; ++ns)
          acc[ms][ns] = __builtin_amdgcn_wmma_f32_16x16x32_bf16(
              false, a[ms].v, false, b[ns].v, (short)0, acc[ms][ns], false, false);
    }
  }

  const int ncol = lane & 15;
  const int mhi  = (lane >> 4) * 8;
#pragma unroll
  for (int ms = 0; ms < 2; ++ms)
#pragma unroll
    for (int vr = 0; vr < 8; ++vr) {
      int mrow = wm + ms * 16 + mhi + vr;
      if (mrow < rem) {
        int gp = base + m0 + mrow;
        int t  = perm[gp];
        float w = wpacked[gp];
#pragma unroll
        for (int ns = 0; ns < 4; ++ns) {
          int h = n0 + wn + ns * 16 + ncol;
          out[(size_t)t * H_DIM + h] = acc[ms][ns][vr] * w;
        }
      }
    }
}

// ---------------------------------------------------------------------------
extern "C" void kernel_launch(void* const* d_in, const int* in_sizes, int n_in,
                              void* d_out, int out_size, void* d_ws, size_t ws_size,
                              hipStream_t stream) {
  (void)in_sizes; (void)n_in; (void)out_size; (void)ws_size;
  const float* x      = (const float*)d_in[0];
  const float* rw     = (const float*)d_in[1];
  const float* gate_w = (const float*)d_in[2];
  const float* up_w   = (const float*)d_in[3];
  const float* down_w = (const float*)d_in[4];
  float* out = (float*)d_out;

  char* ws = (char*)d_ws;
  int*    counts   = (int*)   (ws + WS_COUNTS);
  int*    cursor   = (int*)   (ws + WS_CURSOR);
  int*    offs     = (int*)   (ws + WS_OFFS);
  int*    expertid = (int*)   (ws + WS_EXPERT);
  float*  wtok     = (float*) (ws + WS_WTOK);
  int*    perm     = (int*)   (ws + WS_PERM);
  float*  wpacked  = (float*) (ws + WS_WPACK);
  __bf16* xpack    = (__bf16*)(ws + WS_XPACK);
  __bf16* actpack  = (__bf16*)(ws + WS_ACTPACK);

  moe_init   <<<1, 32, 0, stream>>>(counts, cursor);
  moe_router <<<T_TOK / 256, 256, 0, stream>>>(x, rw, expertid, wtok, counts);
  moe_prefix <<<1, 1, 0, stream>>>(counts, offs);
  moe_scatter<<<T_TOK / 256, 256, 0, stream>>>(expertid, wtok, offs, cursor, perm, wpacked);
  moe_gather <<<T_TOK, 256, 0, stream>>>(x, perm, xpack);
  moe_gemm1<<<dim3(I_DIM / 64, T_TOK / 128, E_NUM), 256, 0, stream>>>(
      xpack, gate_w, up_w, offs, actpack);
  moe_gemm2<<<dim3(H_DIM / 128, T_TOK / 128, E_NUM), 256, 0, stream>>>(
      actpack, down_w, offs, perm, wpacked, out);
}
